// MultiHeadSelfAttention_46797963657816
// MI455X (gfx1250) — matrix operand order repaired and checked by reading, not compiled, not run on hardware
//
#include <hip/hip_runtime.h>
#include <hip/hip_bf16.h>
#include <math.h>

typedef _Float16 half_t;
typedef __attribute__((ext_vector_type(16))) _Float16 v16h;
typedef __attribute__((ext_vector_type(8)))  _Float16 v8h;
typedef __attribute__((ext_vector_type(8)))  float    v8f;

#define NTOK   4096
#define DMODEL 768
#define NHEAD  12
#define HD     64
#define D3     2304   // 3*DMODEL

union AFrag { v16h v; v8h h[2]; };

static __device__ __forceinline__ int lane_id() { return threadIdx.x & 31; }

// Load A fragment (16x32 f16, M x K) from row-major src (leading dim ld, elems),
// tile origin (row0, k0). CDNA5 16-bit A layout:
//   lanes 0-15 : M=lane,    elems 0-7 -> K=k0+0..7,   elems 8-15 -> K=k0+16..23
//   lanes 16-31: M=lane-16, elems 0-7 -> K=k0+8..15,  elems 8-15 -> K=k0+24..31
static __device__ __forceinline__ v16h load_a_frag(const half_t* src, int ld, int row0, int k0) {
  int lane = lane_id();
  int m = lane & 15, hi = lane >> 4;
  const half_t* p = src + (size_t)(row0 + m) * ld + k0 + hi * 8;
  AFrag f;
  f.h[0] = *(const v8h*)(p);
  f.h[1] = *(const v8h*)(p + 16);
  return f.v;
}

// Load B fragment (32x16 f16, K x N) from N-major storage bt[n][k] (leading dim ld).
// CDNA5 16-bit B layout: lane holds column N=lane&15, K split like A:
//   lanes 0-15: K=k0+0..7,16..23, lanes 16-31: K=k0+8..15,24..31
static __device__ __forceinline__ v16h load_b_frag(const half_t* bt, int ld, int k0, int n0) {
  int lane = lane_id();
  int n = lane & 15, hi = lane >> 4;
  const half_t* p = bt + (size_t)(n0 + n) * ld + k0 + hi * 8;
  AFrag f;
  f.h[0] = *(const v8h*)(p);
  f.h[1] = *(const v8h*)(p + 16);
  return f.v;
}

static __device__ __forceinline__ v8f wmma_f16(v16h a, v16h b, v8f c) {
  return __builtin_amdgcn_wmma_f32_16x16x32_f16(false, a, false, b, (short)0, c, false, false);
}

// ---------------- conversion kernels ----------------

__global__ void cvt_f32_to_f16(const float* __restrict__ in, half_t* __restrict__ out, int n) {
  int i = blockIdx.x * blockDim.x + threadIdx.x;
  if (i < n) out[i] = (half_t)in[i];
}

// in: [K][C] f32 row-major -> out: [C][K] f16 (transposed)
__global__ void cvt_transpose(const float* __restrict__ in, half_t* __restrict__ out, int K, int C) {
  int i = blockIdx.x * blockDim.x + threadIdx.x;
  if (i >= K * C) return;
  int c = i / K, k = i - c * K;
  out[i] = (half_t)in[(size_t)k * C + c];
}

// ---------------- QKV projection ----------------
// qkv = xh[4096x768] @ W_qkv  (via wqkvt = W_qkv^T, [2304][768] f16)
// Each wave: one 32x64 output tile (8 WMMAs per k-step). The 64-col group is
// exactly one head of Q, K or V.
// Q -> q[h][n][d], K -> k[h][n][d], V -> vt[h][d][n] (transposed for PV B-frags).
__global__ void __launch_bounds__(256) qkv_gemm(const half_t* __restrict__ xh,
                                                const half_t* __restrict__ wqkvt,
                                                half_t* __restrict__ qb,
                                                half_t* __restrict__ kbuf,
                                                half_t* __restrict__ vtb) {
  const int TM = NTOK / 32;                       // 128 row tiles (32 rows each)
  int wid = blockIdx.x * 8 + (threadIdx.x >> 5);  // 0..4607
  int tm = wid % TM;
  int tn = wid / TM;                              // 0..35 (64-col tile)
  int row0 = tm * 32;
  v8f acc[2][4] = {};
  for (int kk = 0; kk < DMODEL; kk += 32) {
    v16h a0 = load_a_frag(xh, DMODEL, row0,      kk);
    v16h a1 = load_a_frag(xh, DMODEL, row0 + 16, kk);
#pragma unroll
    for (int j = 0; j < 4; ++j) {
      v16h b = load_b_frag(wqkvt, DMODEL, kk, tn * 64 + j * 16);
      acc[0][j] = wmma_f16(a0, b, acc[0][j]);
      acc[1][j] = wmma_f16(a1, b, acc[1][j]);
    }
  }
  int lane = lane_id(), n = lane & 15, hi = lane >> 4;
  int seg = tn / NHEAD;          // 0=Q, 1=K, 2=V
  int h   = tn % NHEAD;
#pragma unroll
  for (int i = 0; i < 2; ++i) {
#pragma unroll
    for (int j = 0; j < 4; ++j) {
      int d = j * 16 + n;
#pragma unroll
      for (int r = 0; r < 8; ++r) {
        int row = row0 + i * 16 + hi * 8 + r;
        half_t hv = (half_t)acc[i][j][r];
        if (seg == 0)      qb  [((size_t)h * NTOK + row) * HD + d] = hv;
        else if (seg == 1) kbuf[((size_t)h * NTOK + row) * HD + d] = hv;
        else               vtb [((size_t)h * HD + d) * NTOK + row] = hv;
      }
    }
  }
}

// ---------------- flash attention ----------------
// One wave per (head, 16-query tile). 64 keys per iteration:
//   8 score WMMAs + softmax (one shuffle-butterfly per row per 64 keys,
//   exp2-domain) + 8 PV WMMAs.
__global__ void __launch_bounds__(256) attn_kernel(const half_t* __restrict__ qb,
                                                   const half_t* __restrict__ kbuf,
                                                   const half_t* __restrict__ vtb,
                                                   half_t* __restrict__ ao) {
  __shared__ __align__(16) half_t plds[8][16 * 64];  // per-wave P staging (2 KB each)
  int wslot = threadIdx.x >> 5;
  int wid = blockIdx.x * 8 + wslot;                  // 0..3071
  int qt = wid % (NTOK / 16);
  int h  = wid / (NTOK / 16);
  const half_t* qh = qb   + (size_t)h * NTOK * HD;   // [n][d]
  const half_t* kh = kbuf + (size_t)h * NTOK * HD;   // [n][d]  (== (K^T) N-major)
  const half_t* vh = vtb  + (size_t)h * HD * NTOK;   // [d][n]  (== V N-major)
  int row0 = qt * 16;
  v16h aq0 = load_a_frag(qh, HD, row0, 0);
  v16h aq1 = load_a_frag(qh, HD, row0, 32);
  const float sc = 0.125f * 1.44269504088896f;       // 64^-0.5 * log2(e)  (exp2 domain)
  float mi[8], li[8];
#pragma unroll
  for (int r = 0; r < 8; ++r) { mi[r] = -1e30f; li[r] = 0.f; }
  v8f acc[4] = {};
  int lane = lane_id(), n = lane & 15, hi = lane >> 4;
  half_t* pl = &plds[wslot][0];

  for (int kb0 = 0; kb0 < NTOK; kb0 += 64) {
    // S tile: 16 queries x 64 keys, contraction over hd=64 (2 wmma steps per group)
    v8f s[4] = {};
#pragma unroll
    for (int g = 0; g < 4; ++g) {
      s[g] = wmma_f16(aq0, load_b_frag(kh, HD, 0,  kb0 + g * 16), s[g]);
      s[g] = wmma_f16(aq1, load_b_frag(kh, HD, 32, kb0 + g * 16), s[g]);
    }

    float p[4][8], alpha[8];
#pragma unroll
    for (int r = 0; r < 8; ++r) {
      float a0 = s[0][r] * sc, a1 = s[1][r] * sc;
      float a2 = s[2][r] * sc, a3 = s[3][r] * sc;
      float mx = fmaxf(fmaxf(a0, a1), fmaxf(a2, a3)); // row max across 16 lanes
      mx = fmaxf(mx, __shfl_xor(mx, 1));
      mx = fmaxf(mx, __shfl_xor(mx, 2));
      mx = fmaxf(mx, __shfl_xor(mx, 4));
      mx = fmaxf(mx, __shfl_xor(mx, 8));
      float nm = fmaxf(mi[r], mx);
      alpha[r] = exp2f(mi[r] - nm);
      p[0][r] = exp2f(a0 - nm);
      p[1][r] = exp2f(a1 - nm);
      p[2][r] = exp2f(a2 - nm);
      p[3][r] = exp2f(a3 - nm);
      float rs = (p[0][r] + p[1][r]) + (p[2][r] + p[3][r]);
      rs += __shfl_xor(rs, 1);
      rs += __shfl_xor(rs, 2);
      rs += __shfl_xor(rs, 4);
      rs += __shfl_xor(rs, 8);
      li[r] = li[r] * alpha[r] + rs;
      mi[r] = nm;
    }
#pragma unroll
    for (int j = 0; j < 4; ++j)
#pragma unroll
      for (int r = 0; r < 8; ++r) acc[j][r] *= alpha[r];

    // Re-layout P (C-fragment f32, 16x64) -> two A-fragments f16 via LDS
#pragma unroll
    for (int g = 0; g < 4; ++g)
#pragma unroll
      for (int r = 0; r < 8; ++r)
        pl[(hi * 8 + r) * 64 + g * 16 + n] = (half_t)p[g][r];
    asm volatile("s_wait_dscnt 0" ::: "memory");
    v16h ap0 = load_a_frag(pl, 64, 0, 0);
    v16h ap1 = load_a_frag(pl, 64, 0, 32);

    // O += P(16x64) @ V(64x64), contraction over keys in 2 wmma steps
#pragma unroll
    for (int j = 0; j < 4; ++j)
      acc[j] = wmma_f16(ap0, load_b_frag(vh, NTOK, kb0, j * 16), acc[j]);
#pragma unroll
    for (int j = 0; j < 4; ++j)
      acc[j] = wmma_f16(ap1, load_b_frag(vh, NTOK, kb0 + 32, j * 16), acc[j]);
    asm volatile("s_wait_dscnt 0" ::: "memory");  // LDS reused next iteration
  }

  float inv[8];
#pragma unroll
  for (int r = 0; r < 8; ++r) inv[r] = 1.0f / li[r];
#pragma unroll
  for (int j = 0; j < 4; ++j) {
    int d = j * 16 + n;
#pragma unroll
    for (int r = 0; r < 8; ++r) {
      int row = row0 + hi * 8 + r;
      ao[(size_t)row * DMODEL + h * HD + d] = (half_t)(acc[j][r] * inv[r]);
    }
  }
}

// ---------------- output projection ----------------
// 32x64 tile per wave, f32 epilogue with bias.
__global__ void __launch_bounds__(256) out_gemm(const half_t* __restrict__ ao,
                                                const half_t* __restrict__ wot,
                                                const float* __restrict__ bias,
                                                float* __restrict__ out) {
  const int TM = NTOK / 32;                       // 128
  int wid = blockIdx.x * 8 + (threadIdx.x >> 5);  // 0..1535
  int tm = wid % TM;
  int tn = wid / TM;                              // 0..11
  int row0 = tm * 32;
  v8f acc[2][4] = {};
  for (int kk = 0; kk < DMODEL; kk += 32) {
    v16h a0 = load_a_frag(ao, DMODEL, row0,      kk);
    v16h a1 = load_a_frag(ao, DMODEL, row0 + 16, kk);
#pragma unroll
    for (int j = 0; j < 4; ++j) {
      v16h b = load_b_frag(wot, DMODEL, kk, tn * 64 + j * 16);
      acc[0][j] = wmma_f16(a0, b, acc[0][j]);
      acc[1][j] = wmma_f16(a1, b, acc[1][j]);
    }
  }
  int lane = lane_id(), n = lane & 15, hi = lane >> 4;
#pragma unroll
  for (int i = 0; i < 2; ++i) {
#pragma unroll
    for (int j = 0; j < 4; ++j) {
      int col = tn * 64 + j * 16 + n;
      float bv = bias[col];
#pragma unroll
      for (int r = 0; r < 8; ++r) {
        int row = row0 + i * 16 + hi * 8 + r;
        out[(size_t)row * DMODEL + col] = acc[i][j][r] + bv;
      }
    }
  }
}

extern "C" void kernel_launch(void* const* d_in, const int* in_sizes, int n_in,
                              void* d_out, int out_size, void* d_ws, size_t ws_size,
                              hipStream_t stream) {
  (void)in_sizes; (void)n_in; (void)out_size; (void)ws_size;
  const float* x    = (const float*)d_in[0];   // [4096][768]
  const float* wqkv = (const float*)d_in[1];   // [768][2304]
  const float* wout = (const float*)d_in[2];   // [768][768]
  const float* bout = (const float*)d_in[3];   // [768]
  float* out = (float*)d_out;                  // [4096][768]

  half_t* ws = (half_t*)d_ws;
  size_t off = 0;
  half_t* xh    = ws + off; off += (size_t)NTOK * DMODEL;    // x f16
  half_t* wqkvt = ws + off; off += (size_t)D3 * DMODEL;      // W_qkv^T f16
  half_t* wot   = ws + off; off += (size_t)DMODEL * DMODEL;  // W_out^T f16
  half_t* qb    = ws + off; off += (size_t)NHEAD * NTOK * HD;
  half_t* kbuf  = ws + off; off += (size_t)NHEAD * NTOK * HD;
  half_t* vtb   = ws + off; off += (size_t)NHEAD * HD * NTOK;
  half_t* ao    = ws + off; off += (size_t)NTOK * DMODEL;    // attention output f16

  int nx = NTOK * DMODEL;
  cvt_f32_to_f16<<<(nx + 255) / 256, 256, 0, stream>>>(x, xh, nx);
  cvt_transpose<<<(DMODEL * D3 + 255) / 256, 256, 0, stream>>>(wqkv, wqkvt, DMODEL, D3);
  cvt_transpose<<<(DMODEL * DMODEL + 255) / 256, 256, 0, stream>>>(wout, wot, DMODEL, DMODEL);

  qkv_gemm<<<(128 * 36) / 8, 256, 0, stream>>>(xh, wqkvt, qb, kbuf, vtb);
  attn_kernel<<<(NHEAD * 256) / 8, 256, 0, stream>>>(qb, kbuf, vtb, ao);
  out_gemm<<<(128 * 12) / 8, 256, 0, stream>>>(ao, wot, bout, out);
}
